// BinNormTrain_86775519248464
// MI455X (gfx1250) — compile-verified, matrix operand order checked by reading, not run on hardware
//
#include <hip/hip_runtime.h>
#include <hip/hip_bf16.h>

typedef __attribute__((ext_vector_type(16))) _Float16 v16h;
typedef __attribute__((ext_vector_type(8)))  float    v8f;

#define D_DIM      256
#define ROWS_PER_BLK 8
#define K_TARGET   64.0f
#define EPS_W      1e-4f
#define N_ITERS    10
#define DELTA_B    7.0f

// sigmoid(t) = 1/(1 + exp(-t)) = 1/(1 + exp2(-log2(e)*t))
#define NEG_LOG2E  (-1.44269504088896340736f)

#if __has_builtin(__builtin_amdgcn_exp2f)
#define EXP2F(v) __builtin_amdgcn_exp2f(v)
#else
#define EXP2F(v) exp2f(v)
#endif

#if __has_builtin(__builtin_amdgcn_rcpf)
#define RCPF(v) __builtin_amdgcn_rcpf(v)
#else
#define RCPF(v) (1.0f / (v))
#endif

__device__ __forceinline__ unsigned ballot32(bool p) {
#if __has_builtin(__builtin_amdgcn_ballot_w32)
  return __builtin_amdgcn_ballot_w32(p);
#else
  return (unsigned)__ballot(p);
#endif
}

// sigmoid of (x + nu) given anu = nu * NEG_LOG2E
#define SIGH(xv) ((_Float16)RCPF(1.0f + EXP2F(fmaf((xv), NEG_LOG2E, anu))))

__global__ void __launch_bounds__(256)
binnorm_kernel(const float* __restrict__ x, float* __restrict__ out, int rows) {
  __shared__ __align__(16) float xs[ROWS_PER_BLK][D_DIM];

  const int lane = threadIdx.x & 31;
  const int wave = threadIdx.x >> 5;
  const int row  = blockIdx.x * ROWS_PER_BLK + wave;
  const bool active = (row < rows);

  float4 v0, v1;
  if (active) {
    const float4* xr = (const float4*)(x + (size_t)row * D_DIM);
    v0 = xr[lane * 2 + 0];
    v1 = xr[lane * 2 + 1];
    float4* ls = (float4*)xs[wave];
    ls[lane * 2 + 0] = v0;
    ls[lane * 2 + 1] = v1;
  }
  __syncthreads();
  if (!active) return;

  // ---- row min / max -> always-valid initial bracket ----
  float mx = fmaxf(fmaxf(fmaxf(v0.x, v0.y), fmaxf(v0.z, v0.w)),
                   fmaxf(fmaxf(v1.x, v1.y), fmaxf(v1.z, v1.w)));
  float mn = fminf(fminf(fminf(v0.x, v0.y), fminf(v0.z, v0.w)),
                   fminf(fminf(v1.x, v1.y), fminf(v1.z, v1.w)));
#pragma unroll
  for (int off = 16; off > 0; off >>= 1) {
    mx = fmaxf(mx, __shfl_xor(mx, off, 32));
    mn = fminf(mn, __shfl_xor(mn, off, 32));
  }
  // f(-max-7) <= 256*sigmoid(-7) - 64 < 0 ; f(-min+7) >= 256*sigmoid(7) - 64 > 0
  float nl  = -mx - DELTA_B;
  float nuU = -mn + DELTA_B;

  // B = ones (32x16 f16): with B==1, D[m,n] = sum_k A[m,k] + C[m,n]
  v16h bones;
#pragma unroll
  for (int t = 0; t < 16; ++t) bones[t] = (_Float16)1.0f;

  const float inv59 = 1.0f / 59.0f;
  const int   mrow  = lane & 15;   // A-matrix row this lane feeds (branch within batch)
  const int   h     = lane >> 4;   // which 16-wide d-slice of each 32-chunk this lane covers
  const float* base = &xs[wave][h * 16];

  float r = nuU - nl;
  for (int it = 0; it < N_ITERS; ++it) {
    r = nuU - nl;                  // faithful: final nu uses pre-update r of last iter
    if (r <= EPS_W) break;         // frozen bracket => identical to reference's masked no-op

    int cnt = 0;                   // #branches with f(nu_j) < 0, over j = 0..59
    for (int bidx = 0; bidx < 4; ++bidx) {
      const int   j   = bidx * 16 + mrow;              // branch index for this lane's A row
      const float nuj = fmaf(r, (float)j * inv59, nl); // nus[j] = r*ls[j] + nl
      const float anu = nuj * NEG_LOG2E;

      v8f c = {};                                      // f32 accumulator (16x16)
#pragma unroll
      for (int cc = 0; cc < 8; ++cc) {                 // eight k=32 chunks of d
        const float4* p = (const float4*)(base + cc * 32);
        float4 q0 = p[0], q1 = p[1], q2 = p[2], q3 = p[3];
        v16h a;
        a[0]  = SIGH(q0.x); a[1]  = SIGH(q0.y); a[2]  = SIGH(q0.z); a[3]  = SIGH(q0.w);
        a[4]  = SIGH(q1.x); a[5]  = SIGH(q1.y); a[6]  = SIGH(q1.z); a[7]  = SIGH(q1.w);
        a[8]  = SIGH(q2.x); a[9]  = SIGH(q2.y); a[10] = SIGH(q2.z); a[11] = SIGH(q2.w);
        a[12] = SIGH(q3.x); a[13] = SIGH(q3.y); a[14] = SIGH(q3.z); a[15] = SIGH(q3.w);
        // D[m,n] += sum_k A[m,k] : row-sum via matmul-by-ones on the matrix pipe
        c = __builtin_amdgcn_wmma_f32_16x16x32_f16(
            /*neg_a=*/false, a, /*neg_b=*/false, bones,
            /*c_mod=*/(short)0, c, /*reuse_a=*/false, /*reuse_b=*/false);
      }

      // C/D layout: VGPR g holds D[g, lane] (lanes 0-15) and D[8+g, lane-16] (lanes 16-31).
      // All columns equal (B==1); read the sign from lanes 0 and 16 via ballot.
#pragma unroll
      for (int g = 0; g < 8; ++g) {
        unsigned m = ballot32(c[g] < K_TARGET);        // fs < 0  <=>  sum < K
        const int j0 = bidx * 16 + g;                  // branch from lane 0 bit
        const int j1 = bidx * 16 + 8 + g;              // branch from lane 16 bit
        cnt += (int)(m & 1u);                          // j0 <= 55 < 60 always valid
        cnt += ((j1 < 60) && ((m >> 16) & 1u)) ? 1 : 0;
      }
    }

    // bracket update, faithful to reference (incl. J-case and JAX index clamp)
    int  i_lo = cnt - 1;
    bool J    = (i_lo < 0);
    if (J) i_lo = 0;
    int  i_hi = i_lo + 1;
    if (i_hi > 59) i_hi = 59;                          // jnp clamps OOB gather index
    float nl2 = fmaf(r, (float)i_lo * inv59, nl);
    float nu2 = fmaf(r, (float)i_hi * inv59, nl);
    if (J) nl2 -= DELTA_B;
    nl  = nl2;
    nuU = nu2;
  }

  const float nu  = fmaf(r, 0.5f, nl);
  const float anu = nu * NEG_LOG2E;

  // out = sigmoid(x + nu)
  float4 o0, o1;
  o0.x = RCPF(1.0f + EXP2F(fmaf(v0.x, NEG_LOG2E, anu)));
  o0.y = RCPF(1.0f + EXP2F(fmaf(v0.y, NEG_LOG2E, anu)));
  o0.z = RCPF(1.0f + EXP2F(fmaf(v0.z, NEG_LOG2E, anu)));
  o0.w = RCPF(1.0f + EXP2F(fmaf(v0.w, NEG_LOG2E, anu)));
  o1.x = RCPF(1.0f + EXP2F(fmaf(v1.x, NEG_LOG2E, anu)));
  o1.y = RCPF(1.0f + EXP2F(fmaf(v1.y, NEG_LOG2E, anu)));
  o1.z = RCPF(1.0f + EXP2F(fmaf(v1.z, NEG_LOG2E, anu)));
  o1.w = RCPF(1.0f + EXP2F(fmaf(v1.w, NEG_LOG2E, anu)));

  float4* orow = (float4*)(out + (size_t)row * D_DIM);
  orow[lane * 2 + 0] = o0;
  orow[lane * 2 + 1] = o1;
}

extern "C" void kernel_launch(void* const* d_in, const int* in_sizes, int n_in,
                              void* d_out, int out_size, void* d_ws, size_t ws_size,
                              hipStream_t stream) {
  const float* x   = (const float*)d_in[0];
  float*       out = (float*)d_out;
  const int rows   = in_sizes[0] / D_DIM;          // 16384
  const int blocks = (rows + ROWS_PER_BLK - 1) / ROWS_PER_BLK;
  binnorm_kernel<<<blocks, 256, 0, stream>>>(x, out, rows);
}
